// UNetBlockOuterSparse_26121991094519
// MI455X (gfx1250) — matrix operand order repaired and checked by reading, not compiled, not run on hardware
//
#include <hip/hip_runtime.h>

typedef __attribute__((ext_vector_type(16))) __bf16 v16bf;
typedef __attribute__((ext_vector_type(8)))  __bf16 v8bf;
typedef __attribute__((ext_vector_type(8)))  float  v8f;

#define CMAX 192   // max channel count that goes through instance-norm

// =====================================================================
// WMMA gather sparse-conv:  Y[M x CoutReal] (+bias) = sum_k gather(X, nmap[:,k]) @ W[k]
//   X  : [* x CinP] bf16 (rows indexed by nmap; CinP multiple of 32)
//   Wt : [K][CoutP][CinP] bf16, pre-transposed + zero padded
//   nmap: [M x K] int (-1 = zero contribution); nullptr = identity
// One wave computes a 16 x (16*NT) output tile via v_wmma_f32_16x16x32_bf16,
// reusing each gathered A-fragment for NT accumulators.
// =====================================================================
template <int NT>
__global__ void k_sconv(const __bf16* __restrict__ X,
                        const __bf16* __restrict__ Wt,
                        const int* __restrict__ nmap,
                        int K, int M, int CinP, int CoutP,
                        const float* __restrict__ bias, int biasC,
                        float* __restrict__ Y, int ldY, int CoutReal)
{
    const int lane  = threadIdx.x & 31;
    const int wave  = threadIdx.x >> 5;
    const int mbase = (blockIdx.x * (blockDim.x >> 5) + wave) * 16;
    if (mbase >= M) return;                     // wave-uniform: EXEC stays all-ones
    const int nbase = blockIdx.y * (16 * NT);
    const int hi = lane >> 4;                   // half-wave selector
    const int la = lane & 15;
    const int arow = mbase + la;                // gathered row owned by this lane (A)
    const bool arow_ok = (arow < M);
    const int nCk = CinP >> 5;                  // 32-wide K chunks
    const size_t wstride_k = (size_t)CoutP * CinP;
    // B^T row for this lane: column nbase+la, K offset hi*16
    const __bf16* wb = Wt + (size_t)(nbase + la) * CinP + hi * 16;

    v8f acc[NT] = {};
    for (int k = 0; k < K; ++k) {
        int r;
        if (nmap) r = arow_ok ? nmap[(size_t)arow * K + k] : -1;
        else      r = arow_ok ? arow : -1;
#if __has_builtin(__builtin_amdgcn_ballot_w32)
        // sparse grids: most taps have no valid neighbor for the whole tile
        if (__builtin_amdgcn_ballot_w32(r >= 0) == 0u) continue;   // uniform
#endif
        const __bf16* xa = X + (size_t)(r < 0 ? 0 : r) * CinP + hi * 16;
        const __bf16* wk = wb + (size_t)k * wstride_k;
        for (int c = 0; c < nCk; ++c) {
            // A fragment: lane<16 -> K {c*32+0..15}; lane>=16 -> {+16..31}
            v16bf a = {};
            if (r >= 0) {
                v8bf lo  = *(const v8bf*)(xa + c * 32);
                v8bf hi8 = *(const v8bf*)(xa + c * 32 + 8);
                a = __builtin_shufflevector(lo, hi8, 0,1,2,3,4,5,6,7,
                                                     8,9,10,11,12,13,14,15);
            }
            #pragma unroll
            for (int t = 0; t < NT; ++t) {
                // B fragment: lane holds col, K = c*32 + hi*16 .. +15 (contiguous)
                v16bf b = *(const v16bf*)(wk + (size_t)t * 16 * CinP + c * 32);
                acc[t] = __builtin_amdgcn_wmma_f32_16x16x32_bf16(false, a, false, b,
                                                                 (short)0, acc[t],
                                                                 false, false);
            }
        }
    }
    #pragma unroll
    for (int t = 0; t < NT; ++t) {
        const int n = nbase + t * 16 + la;
        if (n < CoutReal) {
            float bv = (bias && n < biasC) ? bias[n] : 0.f;
            #pragma unroll
            for (int j = 0; j < 8; ++j) {       // D layout: row = j + 8*hi, col = la
                int row = mbase + hi * 8 + j;
                if (row < M) Y[(size_t)row * ldY + n] = acc[t][j] + bv;
            }
        }
    }
}

// ------------------- instance-norm support -------------------
__global__ void k_zero(float* __restrict__ p, long n) {
    long i = (long)blockIdx.x * blockDim.x + threadIdx.x;
    if (i < n) p[i] = 0.f;
}
__global__ void k_count(const int* __restrict__ bidx, long M, float* __restrict__ cnt) {
    long i = (long)blockIdx.x * blockDim.x + threadIdx.x;
    if (i < M) atomicAdd(&cnt[bidx[i]], 1.f);
}
// acc layout: [0..2C) = sums, [2C..4C) = sumsq
__global__ void k_stats(const float* __restrict__ X, const int* __restrict__ bidx,
                        long M, int C, float* __restrict__ acc)
{
    __shared__ float ss[2 * CMAX];
    __shared__ float sq[2 * CMAX];
    for (int i = threadIdx.x; i < 2 * C; i += blockDim.x) { ss[i] = 0.f; sq[i] = 0.f; }
    __syncthreads();
    long total = M * C;
    for (long i = (long)blockIdx.x * blockDim.x + threadIdx.x; i < total;
         i += (long)gridDim.x * blockDim.x) {
        long m = i / C; int c = (int)(i - m * C);
        float v = X[i];
        int b = bidx[m];
        atomicAdd(&ss[b * C + c], v);
        atomicAdd(&sq[b * C + c], v * v);
    }
    __syncthreads();
    for (int i = threadIdx.x; i < 2 * C; i += blockDim.x) {
        atomicAdd(&acc[i], ss[i]);
        atomicAdd(&acc[2 * C + i], sq[i]);
    }
}
__global__ void k_finalize(const float* __restrict__ acc, const float* __restrict__ cnt,
                           int C, float* __restrict__ mr)
{
    int i = blockIdx.x * blockDim.x + threadIdx.x;
    if (i >= 2 * C) return;
    int b = i / C;
    float n = cnt[b];
    float mean = acc[i] / n;
    float var  = acc[2 * C + i] / n - mean * mean;
    mr[2 * i]     = mean;
    mr[2 * i + 1] = rsqrtf(var + 1e-5f);
}
__global__ void k_apply(const float* __restrict__ X, const int* __restrict__ bidx,
                        const float* __restrict__ mr, long M, int C,
                        const float* __restrict__ residf, const __bf16* __restrict__ residb,
                        int relu, float* __restrict__ outf, __bf16* __restrict__ outb, int ldb)
{
    long i = (long)blockIdx.x * blockDim.x + threadIdx.x;
    long total = M * C;
    if (i >= total) return;
    long m = i / C; int c = (int)(i - m * C);
    int b = bidx[m];
    float v = (X[i] - mr[2 * (b * C + c)]) * mr[2 * (b * C + c) + 1];
    if (residf)      v += residf[i];
    else if (residb) v += (float)residb[(size_t)m * C + c];
    if (relu) v = fmaxf(v, 0.f);
    if (outf) outf[i] = v;
    if (outb) outb[(size_t)m * ldb + c] = (__bf16)v;
}

// ------------------- prep / conversion kernels -------------------
// W [K][Cin][Cout] f32  ->  Wt [K][CoutP][CinP] bf16 (zero padded)
__global__ void k_prep_w(const float* __restrict__ W, __bf16* __restrict__ Wt,
                         int K, int Cin, int Cout, int CinP, int CoutP)
{
    long i = (long)blockIdx.x * blockDim.x + threadIdx.x;
    long total = (long)K * CoutP * CinP;
    if (i >= total) return;
    int ci = (int)(i % CinP);
    long t = i / CinP;
    int co = (int)(t % CoutP);
    int k  = (int)(t / CoutP);
    float v = (ci < Cin && co < Cout) ? W[((long)k * Cin + ci) * Cout + co] : 0.f;
    Wt[i] = (__bf16)v;
}
// W_t [8][229][64] f32 -> Wt [512][256] bf16 : Wt[k*64+o][c] = W_t[k][c][o]
__global__ void k_prep_wt_final(const float* __restrict__ W, __bf16* __restrict__ Wt)
{
    long i = (long)blockIdx.x * blockDim.x + threadIdx.x;
    if (i >= 512L * 256) return;
    int cp = (int)(i & 255);
    int q  = (int)(i >> 8);
    int k = q >> 6, o = q & 63;
    float v = (cp < 229) ? W[((long)k * 229 + cp) * 64 + o] : 0.f;
    Wt[i] = (__bf16)v;
}
__global__ void k_cvt_f2b(const float* __restrict__ src, int lds, int C,
                          __bf16* __restrict__ dst, int ldd, int P, long M)
{
    long i = (long)blockIdx.x * blockDim.x + threadIdx.x;
    if (i >= M * P) return;
    long m = i / P; int p = (int)(i - m * P);
    float v = (p < C) ? src[m * lds + p] : 0.f;
    dst[m * ldd + p] = (__bf16)v;
}
__global__ void k_cvt_b2b(const __bf16* __restrict__ src, int lds, int C,
                          __bf16* __restrict__ dst, int ldd, long M)
{
    long i = (long)blockIdx.x * blockDim.x + threadIdx.x;
    if (i >= M * C) return;
    long m = i / C; int c = (int)(i - m * C);
    dst[m * ldd + c] = src[m * lds + c];
}
// proxy = sparse @ W_occ + b_occ ; cat = [enc, mask ? sparse : 0]
__global__ void k_proxy_cat(const float* __restrict__ sparse, const float* __restrict__ Wocc,
                            const float* __restrict__ bocc, const __bf16* __restrict__ encbf,
                            long N2, float* __restrict__ proxy, __bf16* __restrict__ catbf)
{
    long m = (long)blockIdx.x * blockDim.x + threadIdx.x;
    if (m >= N2) return;
    float p = bocc[0];
    for (int c = 0; c < 128; ++c) p += sparse[m * 128 + c] * Wocc[c];
    proxy[m] = p;
    bool mask = p > 0.f;                         // sigmoid(p) > 0.5
    for (int c = 0; c < 64; ++c)  catbf[m * 192 + c] = encbf[m * 64 + c];
    for (int c = 0; c < 128; ++c) catbf[m * 192 + 64 + c] =
        mask ? (__bf16)sparse[m * 128 + c] : (__bf16)0.f;
}

// =====================================================================
extern "C" void kernel_launch(void* const* d_in, const int* in_sizes, int n_in,
                              void* d_out, int out_size, void* d_ws, size_t ws_size,
                              hipStream_t stream)
{
    const float* features = (const float*)d_in[0];
    const int* nmap1 = (const int*)d_in[1];
    const int* cmap  = (const int*)d_in[2];
    const int* nmap2 = (const int*)d_in[3];
    const int* b1 = (const int*)d_in[4];
    const int* b2 = (const int*)d_in[5];
    const int* b3 = (const int*)d_in[6];
    const float* W_d1 = (const float*)d_in[7];
    const float* W_d2 = (const float*)d_in[8];
    const float* Wd_d = (const float*)d_in[9];
    const float* W_f1 = (const float*)d_in[11];
    const float* W_f2 = (const float*)d_in[12];
    const float* Wd_f = (const float*)d_in[13];
    const float* W_i1 = (const float*)d_in[15];
    const float* W_i2 = (const float*)d_in[16];
    const float* Wd_i = (const float*)d_in[17];
    const float* W_e1a = (const float*)d_in[19];
    const float* W_e1b = (const float*)d_in[20];
    const float* W_ed  = (const float*)d_in[21];
    const float* W_e2a = (const float*)d_in[23];
    const float* W_e2b = (const float*)d_in[24];
    const float* W_sub = (const float*)d_in[25];
    const float* b_sub = (const float*)d_in[26];
    const float* W_occ = (const float*)d_in[27];
    const float* b_occ = (const float*)d_in[28];
    const float* W_pi1 = (const float*)d_in[29];
    const float* W_pi2 = (const float*)d_in[30];
    const float* W_pi3 = (const float*)d_in[31];
    const float* b_pi3 = (const float*)d_in[32];
    const float* W_ps1 = (const float*)d_in[33];
    const float* W_ps2 = (const float*)d_in[34];
    const float* W_ps3 = (const float*)d_in[35];
    const float* b_ps3 = (const float*)d_in[36];
    const float* W_t   = (const float*)d_in[37];

    const long N1 = in_sizes[4];   // b1 length (fine voxels)
    const long N2 = in_sizes[5];   // b2 length (coarse voxels)
    const long N3 = in_sizes[6];   // b3 length = 8*N2

    // ---- output sections (all f32, concatenated) ----
    float* out_main  = (float*)d_out;               // [N3, 64] == [N2, 512]
    float* proxy_out = out_main + (size_t)N3 * 64;  // [N2]
    float* pi_out    = proxy_out + N2;              // [N2, 17]
    float* ps_out    = pi_out + (size_t)N2 * 17;    // [N2, 20]

    // ---- workspace bump allocator ----
    char* wsB = (char*)d_ws;
    size_t off = 0;
    auto alloc = [&](size_t bytes) -> void* {
        void* p = wsB + off;
        off += (bytes + 255) & ~(size_t)255;
        return p;
    };
    auto allocW = [&](long elems) { return (__bf16*)alloc((size_t)elems * 2); };

    // transposed/padded bf16 weights
    __bf16* Wt_d1 = allocW(27L*64*32);
    __bf16* Wt_d2 = allocW(27L*64*64);
    __bf16* WtD_d = allocW(64L*32);
    __bf16* Wt_f1 = allocW(27L*64*96);
    __bf16* Wt_f2 = allocW(27L*64*64);
    __bf16* WtD_f = allocW(64L*96);
    __bf16* Wt_i1 = allocW(27L*64*32);
    __bf16* Wt_i2 = allocW(27L*64*64);
    __bf16* WtD_i = allocW(64L*32);
    __bf16* Wt_e1a = allocW(8L*64*192);
    __bf16* Wt_e1b = allocW(27L*64*64);
    __bf16* Wt_ed  = allocW(8L*64*192);
    __bf16* Wt_e2a = allocW(27L*64*64);
    __bf16* Wt_e2b = allocW(27L*64*64);
    __bf16* Wt_sub = allocW(128L*64);
    __bf16* Wt_pi1 = allocW(27L*192*192);
    __bf16* Wt_pi2 = allocW(27L*192*192);
    __bf16* Wt_pi3 = allocW(27L*32*192);
    __bf16* Wt_ps1 = allocW(27L*192*192);
    __bf16* Wt_ps2 = allocW(27L*192*192);
    __bf16* Wt_ps3 = allocW(27L*32*192);
    __bf16* Wt_t   = allocW(512L*256);

    float* statAcc = (float*)alloc(4 * CMAX * sizeof(float));
    float* mr      = (float*)alloc(4 * CMAX * sizeof(float));
    float* cntbuf  = (float*)alloc(8 * sizeof(float));
    float* cnt1 = cntbuf, *cnt2 = cntbuf + 2, *cnt3 = cntbuf + 4;

    // big buffers (with reuse)
    size_t xp_bytes = (size_t)N1 * 96 * 2;          // padded branch input (bf16)
    size_t tc_bytes = (size_t)N2 * 192 * 4;         // 192-ch conv tmp (f32)
    char*  XpTc = (char*)alloc(xp_bytes > tc_bytes ? xp_bytes : tc_bytes);
    __bf16* Xp = (__bf16*)XpTc;
    float*  tc = (float*)XpTc;                       // level-2 only (Xp dead by then)
    float*  t1   = (float*)alloc((size_t)N1 * 64 * 4);
    float*  resf = (float*)alloc((size_t)N1 * 64 * 4);   // directly follows t1
    float*  sparse = t1;                             // [N2,128] spans t1+resf after both dead
    __bf16* h64  = allocW(N1 * 64);
    __bf16* xbf  = allocW(N1 * 192);                 // concat(e_d,e_f,e_i)
    __bf16* catbf = xbf;                             // reuse: xbf dead before cat built
    __bf16* h192 = allocW(N2 * 192);
    __bf16* encbf = allocW(N2 * 64);
    __bf16* full  = allocW(N2 * 256);                // [N2,256] padded from 229
    (void)ws_size; (void)n_in; (void)out_size;

    auto g1 = [](long n) { return dim3((unsigned)((n + 255) / 256)); };

    auto conv = [&](const __bf16* X, const __bf16* Wt, const int* nm, int K, long M,
                    int CinP, int CoutP, const float* bias, int biasC,
                    float* Y, int ldY, int CoutReal) {
        int NT = (CoutP % 64 == 0) ? 4 : ((CoutP % 32 == 0) ? 2 : 1);
        dim3 grid((unsigned)((M + 63) / 64), (unsigned)(CoutP / (16 * NT)));
        switch (NT) {
        case 4:
            k_sconv<4><<<grid, 128, 0, stream>>>(X, Wt, nm, K, (int)M, CinP, CoutP,
                                                 bias, biasC, Y, ldY, CoutReal);
            break;
        case 2:
            k_sconv<2><<<grid, 128, 0, stream>>>(X, Wt, nm, K, (int)M, CinP, CoutP,
                                                 bias, biasC, Y, ldY, CoutReal);
            break;
        default:
            k_sconv<1><<<grid, 128, 0, stream>>>(X, Wt, nm, K, (int)M, CinP, CoutP,
                                                 bias, biasC, Y, ldY, CoutReal);
            break;
        }
    };
    auto norm = [&](const float* X, const int* bidx, const float* cnt, long M, int C,
                    const float* residf, const __bf16* residb, int relu,
                    float* outf, __bf16* outb, int ldb) {
        k_zero<<<g1(4 * C), 256, 0, stream>>>(statAcc, 4 * C);
        long total = M * (long)C;
        long want = (total + 255) / 256;
        int blocks = (int)(want < 4096 ? want : 4096);
        k_stats<<<blocks, 256, 0, stream>>>(X, bidx, M, C, statAcc);
        k_finalize<<<g1(2 * C), 256, 0, stream>>>(statAcc, cnt, C, mr);
        k_apply<<<g1(total), 256, 0, stream>>>(X, bidx, mr, M, C,
                                               residf, residb, relu, outf, outb, ldb);
    };
    auto prep = [&](const float* W, __bf16* Wt, int K, int Cin, int Cout,
                    int CinP, int CoutP) {
        long total = (long)K * CoutP * CinP;
        k_prep_w<<<g1(total), 256, 0, stream>>>(W, Wt, K, Cin, Cout, CinP, CoutP);
    };

    // ---- weight prep (bias terms followed by instance-norm cancel exactly) ----
    prep(W_d1, Wt_d1, 27, 2, 64, 32, 64);
    prep(W_d2, Wt_d2, 27, 64, 64, 64, 64);
    prep(Wd_d, WtD_d, 1, 2, 64, 32, 64);
    prep(W_f1, Wt_f1, 27, 80, 64, 96, 64);
    prep(W_f2, Wt_f2, 27, 64, 64, 64, 64);
    prep(Wd_f, WtD_f, 1, 80, 64, 96, 64);
    prep(W_i1, Wt_i1, 27, 17, 64, 32, 64);
    prep(W_i2, Wt_i2, 27, 64, 64, 64, 64);
    prep(Wd_i, WtD_i, 1, 17, 64, 32, 64);
    prep(W_e1a, Wt_e1a, 8, 192, 64, 192, 64);
    prep(W_e1b, Wt_e1b, 27, 64, 64, 64, 64);
    prep(W_ed,  Wt_ed,  8, 192, 64, 192, 64);
    prep(W_e2a, Wt_e2a, 27, 64, 64, 64, 64);
    prep(W_e2b, Wt_e2b, 27, 64, 64, 64, 64);
    prep(W_sub, Wt_sub, 1, 64, 128, 64, 128);
    prep(W_pi1, Wt_pi1, 27, 192, 192, 192, 192);
    prep(W_pi2, Wt_pi2, 27, 192, 192, 192, 192);
    prep(W_pi3, Wt_pi3, 27, 192, 17, 192, 32);
    prep(W_ps1, Wt_ps1, 27, 192, 192, 192, 192);
    prep(W_ps2, Wt_ps2, 27, 192, 192, 192, 192);
    prep(W_ps3, Wt_ps3, 27, 192, 20, 192, 32);
    k_prep_wt_final<<<g1(512L * 256), 256, 0, stream>>>(W_t, Wt_t);

    // ---- per-batch counts ----
    k_zero<<<1, 32, 0, stream>>>(cntbuf, 8);
    k_count<<<g1(N1), 256, 0, stream>>>(b1, N1, cnt1);
    k_count<<<g1(N2), 256, 0, stream>>>(b2, N2, cnt2);
    k_count<<<g1(N3), 256, 0, stream>>>(b3, N3, cnt3);

    // ---- level 1: three sparse basic blocks ----
    auto branch = [&](int coff, int C, int P, const __bf16* W1, const __bf16* W2,
                      const __bf16* WD, int xcol) {
        k_cvt_f2b<<<g1(N1 * P), 256, 0, stream>>>(features + coff, 99, C, Xp, P, P, N1);
        conv(Xp, WD, nullptr, 1, N1, P, 64, nullptr, 0, t1, 64, 64);
        norm(t1, b1, cnt1, N1, 64, nullptr, nullptr, 0, resf, nullptr, 0);
        conv(Xp, W1, nmap1, 27, N1, P, 64, nullptr, 0, t1, 64, 64);
        norm(t1, b1, cnt1, N1, 64, nullptr, nullptr, 1, nullptr, h64, 64);
        conv(h64, W2, nmap1, 27, N1, 64, 64, nullptr, 0, t1, 64, 64);
        norm(t1, b1, cnt1, N1, 64, resf, nullptr, 1, nullptr, xbf + xcol, 192);
    };
    branch(0,  2,  32, Wt_d1, Wt_d2, WtD_d, 0);
    branch(2,  80, 96, Wt_f1, Wt_f2, WtD_f, 64);
    branch(82, 17, 32, Wt_i1, Wt_i2, WtD_i, 128);

    // ---- level 2 encoder ----
    conv(xbf, Wt_ed, cmap, 8, N2, 192, 64, nullptr, 0, t1, 64, 64);
    norm(t1, b2, cnt2, N2, 64, nullptr, nullptr, 0, resf, nullptr, 0);
    conv(xbf, Wt_e1a, cmap, 8, N2, 192, 64, nullptr, 0, t1, 64, 64);
    norm(t1, b2, cnt2, N2, 64, nullptr, nullptr, 1, nullptr, h64, 64);
    conv(h64, Wt_e1b, nmap2, 27, N2, 64, 64, nullptr, 0, t1, 64, 64);
    norm(t1, b2, cnt2, N2, 64, resf, nullptr, 1, nullptr, encbf, 64);
    // sbb e2 (residual = enc itself)
    conv(encbf, Wt_e2a, nmap2, 27, N2, 64, 64, nullptr, 0, t1, 64, 64);
    norm(t1, b2, cnt2, N2, 64, nullptr, nullptr, 1, nullptr, h64, 64);
    conv(h64, Wt_e2b, nmap2, 27, N2, 64, 64, nullptr, 0, t1, 64, 64);
    norm(t1, b2, cnt2, N2, 64, nullptr, encbf, 1, nullptr, encbf, 64);

    // ---- sub / proxy / cat ----
    conv(encbf, Wt_sub, nullptr, 1, N2, 64, 128, b_sub, 128, sparse, 128, 128);
    k_proxy_cat<<<g1(N2), 256, 0, stream>>>(sparse, W_occ, b_occ, encbf, N2,
                                            proxy_out, catbf);

    // ---- pi / ps heads ----
    auto head = [&](const __bf16* Wa, const __bf16* Wb, const __bf16* Wc,
                    const float* bias, int CoutR, float* Yout) {
        conv(catbf, Wa, nmap2, 27, N2, 192, 192, nullptr, 0, tc, 192, 192);
        norm(tc, b2, cnt2, N2, 192, nullptr, nullptr, 1, nullptr, h192, 192);
        conv(h192, Wb, nmap2, 27, N2, 192, 192, nullptr, 0, tc, 192, 192);
        norm(tc, b2, cnt2, N2, 192, nullptr, catbf, 1, nullptr, h192, 192);
        conv(h192, Wc, nmap2, 27, N2, 192, 32, bias, CoutR, Yout, CoutR, CoutR);
    };
    head(Wt_pi1, Wt_pi2, Wt_pi3, b_pi3, 17, pi_out);
    head(Wt_ps1, Wt_ps2, Wt_ps3, b_ps3, 20, ps_out);

    // ---- full = concat(cat, pi, ps) padded to 256, bf16 ----
    k_cvt_b2b<<<g1(N2 * 192), 256, 0, stream>>>(catbf, 192, 192, full, 256, N2);
    k_cvt_f2b<<<g1(N2 * 17), 256, 0, stream>>>(pi_out, 17, 17, full + 192, 256, 17, N2);
    k_cvt_f2b<<<g1(N2 * 47), 256, 0, stream>>>(ps_out, 20, 20, full + 209, 256, 47, N2);

    // ---- final einsum GEMM [N2,256]x[256,512] -> d_out, then in-place inorm+relu ----
    conv(full, Wt_t, nullptr, 1, N2, 256, 512, nullptr, 0, out_main, 512, 512);
    norm(out_main, b3, cnt3, N3, 64, nullptr, nullptr, 1, out_main, nullptr, 0);
}
